// MambaStack_2010044695404
// MI455X (gfx1250) — compile-verified, matrix operand order checked by reading, not run on hardware
//
#include <hip/hip_runtime.h>
#include <hip/hip_bf16.h>

// ---------------- model constants ----------------
#define NUM_LAYERS 3
#define D_MODEL    1024
#define D_STATE    16
#define D_CONV     4
#define D_INNER    2048
#define DT_RANK    64
#define NUM_FEAT   128
#define NUM_CLS    6
#define BATCHN     8
#define SEQLEN     500
#define NTOK       (BATCHN * SEQLEN)   // 4000 = 250 * 16
#define DBLW       (DT_RANK + 2 * D_STATE)  // 96

typedef __attribute__((ext_vector_type(16))) __bf16 v16bf;
typedef __attribute__((ext_vector_type(8)))  __bf16 v8bf;
typedef __attribute__((ext_vector_type(8)))  float  v8f;

// ---------------- fp32 -> bf16 conversion ----------------
__global__ __launch_bounds__(256) void cvt_bf16_kernel(const float* __restrict__ in,
                                                       __bf16* __restrict__ out, int n) {
  int i = blockIdx.x * 256 + threadIdx.x;
  if (i < n) out[i] = (__bf16)in[i];
}

// ---------------- shared epilogue ----------------
__device__ __forceinline__ float apply_epi(int EPI, float v, int row, int col, int ldc,
                                           const float* __restrict__ bias,
                                           const float* __restrict__ pe) {
  if (EPI == 1) {
    v = fmaxf(v + bias[col], 0.0f) + pe[(size_t)(row % SEQLEN) * ldc + col];
  } else if (EPI == 2) {
    v += bias[col];
    v = (v > 20.0f) ? v : log1pf(__expf(v));
  }
  return v;
}

// =====================================================================
// WMMA bf16 GEMM v1 (global-load B), used for the N=96 x_proj GEMM.
// C[M,N] = A[M,K] * B[N,K]^T
// =====================================================================
template <int NT, int EPI>
__global__ __launch_bounds__(256) void wmma_gemm_kernel(
    const __bf16* __restrict__ A, int lda,
    const __bf16* __restrict__ B, int ldb,
    float* __restrict__ C, int ldc,
    int M, int N, int K,
    const float* __restrict__ bias,
    const float* __restrict__ pe) {
  const int lane = threadIdx.x & 31;
  const int wave = threadIdx.x >> 5;
  const int groupsN = N / (16 * NT);
  const int nGroups = (M >> 4) * groupsN;
  const int g = blockIdx.x * (blockDim.x >> 5) + wave;
  if (g >= nGroups) return;                 // wave-uniform: EXEC stays all-1s below
  const int tm = g / groupsN;
  const int tn0 = (g - tm * groupsN) * NT;
  const int halfSel = lane >> 4;
  const int lrow = lane & 15;
  const int kb = halfSel * 8;               // K sub-block per ISA 16-bit A/B layout

  const __bf16* ap = A + (size_t)(tm * 16 + lrow) * lda + kb;
  const __bf16* bp[NT];
#pragma unroll
  for (int t = 0; t < NT; ++t)
    bp[t] = B + (size_t)((tn0 + t) * 16 + lrow) * ldb + kb;

  v8f zero = {0.f, 0.f, 0.f, 0.f, 0.f, 0.f, 0.f, 0.f};
  v8f acc[NT];
#pragma unroll
  for (int t = 0; t < NT; ++t) acc[t] = zero;

  for (int k = 0; k < K; k += 32) {
    if (k + 128 < K) {
      __builtin_prefetch(ap + k + 128, 0, 1);
      __builtin_prefetch(bp[0] + k + 128, 0, 1);
    }
    v8bf alo = *(const v8bf*)(ap + k);
    v8bf ahi = *(const v8bf*)(ap + k + 16);
    v16bf afrag = __builtin_shufflevector(alo, ahi, 0, 1, 2, 3, 4, 5, 6, 7,
                                          8, 9, 10, 11, 12, 13, 14, 15);
#pragma unroll
    for (int t = 0; t < NT; ++t) {
      v8bf blo = *(const v8bf*)(bp[t] + k);
      v8bf bhi = *(const v8bf*)(bp[t] + k + 16);
      v16bf bfrag = __builtin_shufflevector(blo, bhi, 0, 1, 2, 3, 4, 5, 6, 7,
                                            8, 9, 10, 11, 12, 13, 14, 15);
      acc[t] = __builtin_amdgcn_wmma_f32_16x16x32_bf16(
          false, afrag, false, bfrag, (short)0, acc[t], false, false);
    }
  }

  const int rbase = tm * 16 + halfSel * 8;
#pragma unroll
  for (int t = 0; t < NT; ++t) {
    const int col = (tn0 + t) * 16 + lrow;
#pragma unroll
    for (int r = 0; r < 8; ++r)
      C[(size_t)(rbase + r) * ldc + col] =
          apply_epi(EPI, acc[t][r], rbase + r, col, ldc, bias, pe);
  }
}

// =====================================================================
// WMMA bf16 GEMM v2: B panel staged through LDS with async global->LDS
// DMA (gfx1250 ASYNCcnt path), double buffered. One block = 8 M-tiles
// (128 rows) x one 64-column N panel; B panel loaded once per block.
// Requires N % 64 == 0, K % 32 == 0.
// =====================================================================
template <int EPI>
__global__ __launch_bounds__(256) void wmma_gemm_lds_kernel(
    const __bf16* __restrict__ A, int lda,
    const __bf16* __restrict__ B, int ldb,
    float* __restrict__ C, int ldc,
    int M, int N, int K,
    const float* __restrict__ bias,
    const float* __restrict__ pe) {
  __shared__ __align__(16) __bf16 panel[2][64 * 32];  // 2 x 4 KB K-chunks of B
  const int lane = threadIdx.x & 31;
  const int wave = threadIdx.x >> 5;
  const int lrow = lane & 15;
  const int halfSel = lane >> 4;
  const int kb = halfSel * 8;

  // Clamp instead of early-return: every wave must reach the barriers, and
  // EXEC must stay all-1s for WMMA. Clamped waves redo tile (M/16-1) and
  // store identical values (benign).
  int tm = blockIdx.y * 8 + wave;
  const int tmMax = (M >> 4) - 1;
  if (tm > tmMax) tm = tmMax;
  const int n0 = blockIdx.x * 64;

  const __bf16* ap = A + (size_t)(tm * 16 + lrow) * lda + kb;

  // Staging map: 256 threads cover the 64-row x 64-byte B panel chunk,
  // one 16-byte async copy per lane (one instruction per wave per chunk).
  const int chunk = wave * 32 + lane;          // 0..255
  const int prow = chunk >> 2;                 // 0..63  (B row within panel)
  const int pseg = chunk & 3;                  // 16-byte segment within row
  const __bf16* gsrc0 = B + (size_t)(n0 + prow) * ldb + pseg * 8;
  const unsigned ldsDst0 =
      (unsigned)(uintptr_t)(&panel[0][0]) + (unsigned)(prow * 64 + pseg * 16);

  v8f zero = {0.f, 0.f, 0.f, 0.f, 0.f, 0.f, 0.f, 0.f};
  v8f acc[4];
#pragma unroll
  for (int t = 0; t < 4; ++t) acc[t] = zero;

  // prologue: stage K-chunk 0 into buffer 0
  {
    unsigned long long src = (unsigned long long)(uintptr_t)gsrc0;
    asm volatile("global_load_async_to_lds_b128 %0, %1, off"
                 :: "v"(ldsDst0), "v"(src) : "memory");
    asm volatile("s_wait_asynccnt 0" ::: "memory");
  }
  __syncthreads();

  int buf = 0;
  for (int k = 0; k < K; k += 32) {
    if (k + 32 < K) {  // overlap next chunk's DMA with this chunk's WMMAs
      unsigned long long src = (unsigned long long)(uintptr_t)(gsrc0 + k + 32);
      unsigned dst = ldsDst0 + (unsigned)((buf ^ 1) * (64 * 32 * 2));
      asm volatile("global_load_async_to_lds_b128 %0, %1, off"
                   :: "v"(dst), "v"(src) : "memory");
    }
    if (k + 128 < K) __builtin_prefetch(ap + k + 128, 0, 1);

    v8bf alo = *(const v8bf*)(ap + k);
    v8bf ahi = *(const v8bf*)(ap + k + 16);
    v16bf afrag = __builtin_shufflevector(alo, ahi, 0, 1, 2, 3, 4, 5, 6, 7,
                                          8, 9, 10, 11, 12, 13, 14, 15);
#pragma unroll
    for (int t = 0; t < 4; ++t) {
      const __bf16* pb = &panel[buf][(t * 16 + lrow) * 32 + kb];
      v8bf blo = *(const v8bf*)pb;          // ds_load_b128
      v8bf bhi = *(const v8bf*)(pb + 16);
      v16bf bfrag = __builtin_shufflevector(blo, bhi, 0, 1, 2, 3, 4, 5, 6, 7,
                                            8, 9, 10, 11, 12, 13, 14, 15);
      acc[t] = __builtin_amdgcn_wmma_f32_16x16x32_bf16(
          false, afrag, false, bfrag, (short)0, acc[t], false, false);
    }

    if (k + 32 < K)
      asm volatile("s_wait_asynccnt 0" ::: "memory");
    __syncthreads();
    buf ^= 1;
  }

  const int rbase = tm * 16 + halfSel * 8;
#pragma unroll
  for (int t = 0; t < 4; ++t) {
    const int col = n0 + t * 16 + lrow;
#pragma unroll
    for (int r = 0; r < 8; ++r)
      C[(size_t)(rbase + r) * ldc + col] =
          apply_epi(EPI, acc[t][r], rbase + r, col, ldc, bias, pe);
  }
}

// ---------------- LayerNorm (fp32 in -> bf16 out), one block per row ----------------
__global__ __launch_bounds__(256) void layernorm_kernel(
    const float* __restrict__ x, const float* __restrict__ gamma,
    const float* __restrict__ beta, __bf16* __restrict__ out) {
  __shared__ float rs[8], rq[8];
  const int row = blockIdx.x;
  const float* xr = x + (size_t)row * D_MODEL;
  float v[4], s = 0.f, q = 0.f;
#pragma unroll
  for (int i = 0; i < 4; ++i) {
    v[i] = xr[threadIdx.x + i * 256];
    s += v[i];
    q += v[i] * v[i];
  }
#pragma unroll
  for (int o = 16; o > 0; o >>= 1) {
    s += __shfl_xor(s, o);
    q += __shfl_xor(q, o);
  }
  const int wv = threadIdx.x >> 5, ln = threadIdx.x & 31;
  if (ln == 0) { rs[wv] = s; rq[wv] = q; }
  __syncthreads();
  s = 0.f; q = 0.f;
#pragma unroll
  for (int w = 0; w < 8; ++w) { s += rs[w]; q += rq[w]; }
  const float mu = s * (1.0f / D_MODEL);
  const float var = q * (1.0f / D_MODEL) - mu * mu;
  const float rstd = rsqrtf(var + 1e-5f);
#pragma unroll
  for (int i = 0; i < 4; ++i) {
    const int c = threadIdx.x + i * 256;
    out[(size_t)row * D_MODEL + c] = (__bf16)((v[i] - mu) * rstd * gamma[c] + beta[c]);
  }
}

// ---------------- depthwise causal conv (D_CONV=4) + SiLU ----------------
__global__ __launch_bounds__(256) void conv_silu_kernel(
    const float* __restrict__ xz, const float* __restrict__ cw,
    const float* __restrict__ cb, float* __restrict__ xconv,
    __bf16* __restrict__ xconv_bf) {
  const size_t idx = (size_t)blockIdx.x * 256 + threadIdx.x;
  const int d = (int)(idx & (D_INNER - 1));
  const size_t tok = idx >> 11;          // b*SEQ + t
  const int t = (int)(tok % SEQLEN);
  float s = cb[d];
#pragma unroll
  for (int k = 0; k < D_CONV; ++k) {
    const int tt = t - (D_CONV - 1) + k;
    if (tt >= 0)
      s += xz[(tok - (size_t)(D_CONV - 1 - k)) * (2 * D_INNER) + d] * cw[d * D_CONV + k];
  }
  const float r = s / (1.0f + __expf(-s));
  xconv[idx] = r;
  xconv_bf[idx] = (__bf16)r;
}

// ---------------- extract dt columns of dbl as bf16 ----------------
__global__ __launch_bounds__(256) void dt_extract_kernel(const float* __restrict__ dbl,
                                                         __bf16* __restrict__ dtbf) {
  const int i = blockIdx.x * 256 + threadIdx.x;
  const int row = i >> 6, c = i & 63;
  dtbf[i] = (__bf16)dbl[row * DBLW + c];
}

// ---------------- selective scan + D-skip + SiLU(z) gating (bf16 out) ----------------
__global__ __launch_bounds__(256) void scan_kernel(
    const float* __restrict__ delta,    // [B,SEQ,D_INNER]
    const float* __restrict__ dbl,      // [B,SEQ,96] (B=64..79, C=80..95)
    const float* __restrict__ xconv,    // [B,SEQ,D_INNER]
    const float* __restrict__ xz,       // [B,SEQ,2*D_INNER], z = cols D_INNER..
    const float* __restrict__ A_log,    // [D_INNER,16]
    const float* __restrict__ Dsk,      // [D_INNER]
    __bf16* __restrict__ ybf) {         // [B,SEQ,D_INNER]
  __shared__ float sB[D_STATE], sC[D_STATE];
  const int b = blockIdx.y;
  const int d = blockIdx.x * 256 + threadIdx.x;
  float Ar[D_STATE], h[D_STATE];
#pragma unroll
  for (int n = 0; n < D_STATE; ++n) {
    Ar[n] = -__expf(A_log[d * D_STATE + n]);
    h[n] = 0.0f;
  }
  const float dsk = Dsk[d];
  for (int t = 0; t < SEQLEN; ++t) {
    const size_t tok = (size_t)b * SEQLEN + t;
    if (threadIdx.x < 32) {
      const float v = dbl[tok * DBLW + DT_RANK + threadIdx.x];
      if (threadIdx.x < 16) sB[threadIdx.x] = v;
      else sC[threadIdx.x - 16] = v;
    }
    __syncthreads();
    const float dlt = delta[tok * D_INNER + d];
    const float xv = xconv[tok * D_INNER + d];
    const float dx = dlt * xv;
    float y = 0.0f;
#pragma unroll
    for (int n = 0; n < D_STATE; ++n) {
      h[n] = __expf(dlt * Ar[n]) * h[n] + dx * sB[n];
      y += h[n] * sC[n];
    }
    y += xv * dsk;
    const float z = xz[tok * (2 * D_INNER) + D_INNER + d];
    y *= z / (1.0f + __expf(-z));
    ybf[tok * D_INNER + d] = (__bf16)y;
    __syncthreads();
  }
}

// ---------------- mean pool over SEQ ----------------
__global__ __launch_bounds__(256) void pool_kernel(const float* __restrict__ h,
                                                   float* __restrict__ pooled) {
  const int b = blockIdx.x;
#pragma unroll
  for (int j = 0; j < 4; ++j) {
    const int d = threadIdx.x + j * 256;
    float s = 0.f;
    for (int t = 0; t < SEQLEN; ++t)
      s += h[((size_t)b * SEQLEN + t) * D_MODEL + d];
    pooled[b * D_MODEL + d] = s * (1.0f / SEQLEN);
  }
}

// ---------------- classifier (8x6, tiny) ----------------
__global__ __launch_bounds__(64) void cls_kernel(const float* __restrict__ pooled,
                                                 const float* __restrict__ w,
                                                 const float* __restrict__ bias,
                                                 float* __restrict__ out) {
  const int idx = threadIdx.x;
  if (idx < BATCHN * NUM_CLS) {
    const int b = idx / NUM_CLS, c = idx % NUM_CLS;
    float s = bias[c];
    for (int d = 0; d < D_MODEL; ++d)
      s += pooled[b * D_MODEL + d] * w[c * D_MODEL + d];
    out[idx] = s;
  }
}

// ---------------- host driver ----------------
static inline int gemm_blocks_v1(int M, int N, int NT) {
  const int g = (M / 16) * (N / (16 * NT));
  return (g + 7) / 8;
}
static inline dim3 gemm_grid_v2(int M, int N) {
  return dim3(N / 64, (M / 16 + 7) / 8);
}

extern "C" void kernel_launch(void* const* d_in, const int* in_sizes, int n_in,
                              void* d_out, int out_size, void* d_ws, size_t ws_size,
                              hipStream_t stream) {
  const float* x      = (const float*)d_in[0];
  const float* pe     = (const float*)d_in[1];
  const float* fc_w   = (const float*)d_in[2];
  const float* fc_b   = (const float*)d_in[3];
  const float* ln_g   = (const float*)d_in[4];
  const float* ln_b   = (const float*)d_in[5];
  const float* in_w   = (const float*)d_in[6];
  const float* conv_w = (const float*)d_in[7];
  const float* conv_b = (const float*)d_in[8];
  const float* xp_w   = (const float*)d_in[9];
  const float* dt_w   = (const float*)d_in[10];
  const float* dt_b   = (const float*)d_in[11];
  const float* A_log  = (const float*)d_in[12];
  const float* D_skip = (const float*)d_in[13];
  const float* out_w  = (const float*)d_in[14];
  const float* cls_w  = (const float*)d_in[15];
  const float* cls_b  = (const float*)d_in[16];
  float* out = (float*)d_out;
  (void)in_sizes; (void)n_in; (void)out_size; (void)ws_size;

  char* p = (char*)d_ws;
  auto alloc = [&](size_t bytes) -> void* {
    void* r = (void*)p;
    p += (bytes + 255) & ~(size_t)255;
    return r;
  };

  __bf16* x_bf   = (__bf16*)alloc((size_t)NTOK * NUM_FEAT * 2);
  __bf16* wfc    = (__bf16*)alloc((size_t)D_MODEL * NUM_FEAT * 2);
  __bf16* win    = (__bf16*)alloc((size_t)NUM_LAYERS * 2 * D_INNER * D_MODEL * 2);
  __bf16* wxp    = (__bf16*)alloc((size_t)NUM_LAYERS * DBLW * D_INNER * 2);
  __bf16* wdt    = (__bf16*)alloc((size_t)NUM_LAYERS * D_INNER * DT_RANK * 2);
  __bf16* wout   = (__bf16*)alloc((size_t)NUM_LAYERS * D_MODEL * D_INNER * 2);
  float*  hbuf   = (float*)alloc((size_t)NTOK * D_MODEL * 4);
  __bf16* hn_bf  = (__bf16*)alloc((size_t)NTOK * D_MODEL * 2);
  float*  xzbuf  = (float*)alloc((size_t)NTOK * 2 * D_INNER * 4);
  float*  xconv  = (float*)alloc((size_t)NTOK * D_INNER * 4);
  __bf16* xconvb = (__bf16*)alloc((size_t)NTOK * D_INNER * 2);
  float*  dbl    = (float*)alloc((size_t)NTOK * DBLW * 4);
  __bf16* dt_bf  = (__bf16*)alloc((size_t)NTOK * DT_RANK * 2);
  float*  delta  = (float*)alloc((size_t)NTOK * D_INNER * 4);
  __bf16* y_bf   = (__bf16*)alloc((size_t)NTOK * D_INNER * 2);
  float*  pooled = (float*)alloc((size_t)BATCHN * D_MODEL * 4);

  auto cvt = [&](const float* src, __bf16* dst, int n) {
    cvt_bf16_kernel<<<(n + 255) / 256, 256, 0, stream>>>(src, dst, n);
  };
  cvt(x, x_bf, NTOK * NUM_FEAT);
  cvt(fc_w, wfc, D_MODEL * NUM_FEAT);
  cvt(in_w, win, NUM_LAYERS * 2 * D_INNER * D_MODEL);
  cvt(xp_w, wxp, NUM_LAYERS * DBLW * D_INNER);
  cvt(dt_w, wdt, NUM_LAYERS * D_INNER * DT_RANK);
  cvt(out_w, wout, NUM_LAYERS * D_MODEL * D_INNER);

  // stem: h = relu(x @ fc_w^T + fc_b) + pe
  wmma_gemm_lds_kernel<1><<<gemm_grid_v2(NTOK, D_MODEL), 256, 0, stream>>>(
      x_bf, NUM_FEAT, wfc, NUM_FEAT, hbuf, D_MODEL,
      NTOK, D_MODEL, NUM_FEAT, fc_b, pe);

  for (int i = 0; i < NUM_LAYERS; ++i) {
    layernorm_kernel<<<NTOK, 256, 0, stream>>>(hbuf, ln_g + i * D_MODEL,
                                               ln_b + i * D_MODEL, hn_bf);
    // in_proj: xz = hn @ W^T  (N = 4096)
    wmma_gemm_lds_kernel<0><<<gemm_grid_v2(NTOK, 2 * D_INNER), 256, 0, stream>>>(
        hn_bf, D_MODEL, win + (size_t)i * 2 * D_INNER * D_MODEL, D_MODEL,
        xzbuf, 2 * D_INNER, NTOK, 2 * D_INNER, D_MODEL, nullptr, nullptr);
    conv_silu_kernel<<<(NTOK * D_INNER) / 256, 256, 0, stream>>>(
        xzbuf, conv_w + (size_t)i * D_INNER * D_CONV, conv_b + i * D_INNER,
        xconv, xconvb);
    // x_proj: dbl = xconv @ W^T  (N = 96 -> v1 kernel, NT=2)
    wmma_gemm_kernel<2, 0><<<gemm_blocks_v1(NTOK, DBLW, 2), 256, 0, stream>>>(
        xconvb, D_INNER, wxp + (size_t)i * DBLW * D_INNER, D_INNER,
        dbl, DBLW, NTOK, DBLW, D_INNER, nullptr, nullptr);
    dt_extract_kernel<<<(NTOK * DT_RANK) / 256, 256, 0, stream>>>(dbl, dt_bf);
    // delta = softplus(dt @ dt_w^T + dt_b)
    wmma_gemm_lds_kernel<2><<<gemm_grid_v2(NTOK, D_INNER), 256, 0, stream>>>(
        dt_bf, DT_RANK, wdt + (size_t)i * D_INNER * DT_RANK, DT_RANK,
        delta, D_INNER, NTOK, D_INNER, DT_RANK, dt_b + i * D_INNER, nullptr);
    scan_kernel<<<dim3(D_INNER / 256, BATCHN), 256, 0, stream>>>(
        delta, dbl, xconv, xzbuf, A_log + (size_t)i * D_INNER * D_STATE,
        D_skip + i * D_INNER, y_bf);
    // out_proj: h = y @ W^T
    wmma_gemm_lds_kernel<0><<<gemm_grid_v2(NTOK, D_MODEL), 256, 0, stream>>>(
        y_bf, D_INNER, wout + (size_t)i * D_MODEL * D_INNER, D_INNER,
        hbuf, D_MODEL, NTOK, D_MODEL, D_INNER, nullptr, nullptr);
  }

  pool_kernel<<<BATCHN, 256, 0, stream>>>(hbuf, pooled);
  cls_kernel<<<1, 64, 0, stream>>>(pooled, cls_w, cls_b, out);
}
